// PyramidModuleV2_30253749633638
// MI455X (gfx1250) — compile-verified, matrix operand order checked by reading, not run on hardware
//
#include <hip/hip_runtime.h>
#include <hip/hip_bf16.h>
#include <math.h>

// ---------------------------------------------------------------------------
// Problem constants (from reference)
// ---------------------------------------------------------------------------
#define BATCH   2
#define NPTS    4096
#define CFEAT   64
#define CIN     67      // C + 3
#define CINP    96      // CIN padded to multiple of 32 (WMMA K step)
#define RANCH   128
#define DMODEL  128
#define NHEAD   4
#define HDIM    32
#define NS_PYR  16
#define LN_EPS  1e-5f
#define LDS_PAD 8       // f16 elements of row padding (shifts banks by 4/row)

typedef _Float16 f16;
typedef __attribute__((ext_vector_type(8)))  f16   h8;
typedef __attribute__((ext_vector_type(16))) f16   h16;
typedef __attribute__((ext_vector_type(8)))  float f32x8;

// ---------------------------------------------------------------------------
// Input indexing (setup_inputs() dict order, params flattened in insert order)
// ---------------------------------------------------------------------------
enum {
  I_XYZ = 0, I_FEAT, I_ANCHOR, I_NX0, I_NR0, I_NX1, I_NR1,
  I_P0W1, I_P0W2, I_P1W1, I_P1W2,
  I_PYR0 = 11  // + 32*i per pyramid level
};
enum {
  P_WR1 = 0, P_WR2, P_WPOS, P_WKEY, P_WV1, P_WV2, P_WATTN,
  P_WKC, P_WQC, P_WQKC, P_WVC, P_PE, P_WQKV, P_BQKV, P_WO, P_BO,
  P_W1, P_B1, P_W2, P_B2, P_LN1G, P_LN1B, P_LN2G, P_LN2B,
  P_CW1, P_CB1, P_CW2, P_CB2, P_RW1, P_RB1, P_RW2, P_RB2
};

// ---------------------------------------------------------------------------
// CDNA5 WMMA fragment loader (16-bit A/B operand layout).
// Lanes 0-15 hold row m = lane with K {0..7} in elems 0..7 and K {16..23} in
// elems 8..15; lanes 16-31 hold the same rows with K {8..15} / {24..31}.
// Works for row-major A and for B expressed as Wt = B^T (N x K row-major),
// and for either global or LDS pointers (addrspace inferred after inlining).
// ---------------------------------------------------------------------------
__device__ __forceinline__ h16 frag_rm(const f16* __restrict__ base, int ld, int lane) {
  const f16* p = base + (size_t)(lane & 15) * ld + ((lane >> 4) << 3);
  h8 lo = *(const h8*)(p);       // K offsets {0..7} or {8..15}
  h8 hi = *(const h8*)(p + 16);  // K offsets {16..23} or {24..31}
  return __builtin_shufflevector(lo, hi, 0, 1, 2, 3, 4, 5, 6, 7,
                                         8, 9, 10, 11, 12, 13, 14, 15);
}

// ---------------------------------------------------------------------------
// WMMA GEMM specialized for N == 128:
//   C[M x 128] = act(A[M x K] * Wt[128 x K]^T + bias)
// Block = 8 waves. The block stages Wt into LDS once (row-padded to kill
// bank conflicts); each wave owns a 16-row strip and keeps 8 f32x8
// accumulators, so one global A-fragment feeds 8 back-to-back WMMAs.
// ---------------------------------------------------------------------------
__global__ __launch_bounds__(256) void k_wmma_gemm128(
    const f16* __restrict__ A, int lda,
    const f16* __restrict__ Wt, int ldw,
    const float* __restrict__ bias,
    f16* __restrict__ Ch, float* __restrict__ Cf, int ldc,
    int M, int K, int act)
{
  extern __shared__ f16 sW[];            // [128 x (K + LDS_PAD)]
  const int ldsw = K + LDS_PAD;
  const int lane = threadIdx.x & 31;
  const int wid  = threadIdx.x >> 5;

  // ---- cooperative weight staging: global -> LDS (h8 vectors) ----
  {
    int totalV = (DMODEL * K) >> 3;                 // h8 elements
    for (int v = threadIdx.x; v < totalV; v += blockDim.x) {
      int e   = v << 3;
      int row = e / K;
      int k   = e - row * K;
      *(h8*)(sW + row * ldsw + k) = *(const h8*)(Wt + (size_t)row * ldw + k);
    }
  }
  __syncthreads();

  int tm = blockIdx.x * 8 + wid;                    // 16-row strip index
  if (tm * 16 >= M) return;

  const f16* a = A + (size_t)tm * 16 * lda;

  f32x8 acc0 = {0.f,0.f,0.f,0.f,0.f,0.f,0.f,0.f};
  f32x8 acc1 = acc0, acc2 = acc0, acc3 = acc0;
  f32x8 acc4 = acc0, acc5 = acc0, acc6 = acc0, acc7 = acc0;

  for (int k0 = 0; k0 < K; k0 += 32) {
    __builtin_prefetch(a + k0 + 64, 0, 1);          // global_prefetch_b8
    h16 fa = frag_rm(a + k0, lda, lane);            // one A frag, 8 WMMAs
    h16 fb;
    fb = frag_rm(sW + 0 * 16 * ldsw + k0, ldsw, lane);
    acc0 = __builtin_amdgcn_wmma_f32_16x16x32_f16(false, fa, false, fb, (short)0, acc0, false, false);
    fb = frag_rm(sW + 1 * 16 * ldsw + k0, ldsw, lane);
    acc1 = __builtin_amdgcn_wmma_f32_16x16x32_f16(false, fa, false, fb, (short)0, acc1, false, false);
    fb = frag_rm(sW + 2 * 16 * ldsw + k0, ldsw, lane);
    acc2 = __builtin_amdgcn_wmma_f32_16x16x32_f16(false, fa, false, fb, (short)0, acc2, false, false);
    fb = frag_rm(sW + 3 * 16 * ldsw + k0, ldsw, lane);
    acc3 = __builtin_amdgcn_wmma_f32_16x16x32_f16(false, fa, false, fb, (short)0, acc3, false, false);
    fb = frag_rm(sW + 4 * 16 * ldsw + k0, ldsw, lane);
    acc4 = __builtin_amdgcn_wmma_f32_16x16x32_f16(false, fa, false, fb, (short)0, acc4, false, false);
    fb = frag_rm(sW + 5 * 16 * ldsw + k0, ldsw, lane);
    acc5 = __builtin_amdgcn_wmma_f32_16x16x32_f16(false, fa, false, fb, (short)0, acc5, false, false);
    fb = frag_rm(sW + 6 * 16 * ldsw + k0, ldsw, lane);
    acc6 = __builtin_amdgcn_wmma_f32_16x16x32_f16(false, fa, false, fb, (short)0, acc6, false, false);
    fb = frag_rm(sW + 7 * 16 * ldsw + k0, ldsw, lane);
    acc7 = __builtin_amdgcn_wmma_f32_16x16x32_f16(false, fa, false, fb, (short)0, acc7, false, false);
  }

  // ---- epilogue: C/D layout -> row tm*16 + i (+8 for hi lanes), col lane&15
  int rbase = tm * 16 + ((lane >> 4) << 3);
  int cbase = lane & 15;
  f32x8 accs[8] = {acc0, acc1, acc2, acc3, acc4, acc5, acc6, acc7};
#pragma unroll
  for (int nt = 0; nt < 8; ++nt) {
    int col = nt * 16 + cbase;
    float bv = bias ? bias[col] : 0.f;
#pragma unroll
    for (int i = 0; i < 8; ++i) {
      float v = accs[nt][i] + bv;
      if (act == 1) v = v > 0.f ? v : 0.f;
      size_t off = (size_t)(rbase + i) * ldc + col;
      if (Ch) Ch[off] = (f16)v;
      if (Cf) Cf[off] = v;
    }
  }
}

// ---------------------------------------------------------------------------
// f32 -> f16 weight conversion with K padding
// ---------------------------------------------------------------------------
__global__ void k_cvt(const float* __restrict__ src, f16* __restrict__ dst,
                      int Nrows, int K, int Kp)
{
  int idx = blockIdx.x * blockDim.x + threadIdx.x;
  if (idx >= Nrows * Kp) return;
  int n = idx / Kp, k = idx - n * Kp;
  dst[idx] = (k < K) ? (f16)src[(size_t)n * K + k] : (f16)0.f;
}

// ---------------------------------------------------------------------------
// Ball query + gather for predict_level.  One wave per query; ordered
// first-ns selection via ballot/popc (matches top_k over index order).
// ---------------------------------------------------------------------------
__global__ void k_bq_gather_pred(
    const float* __restrict__ xyz, const float* __restrict__ feats,
    const float* __restrict__ anchor, float r2, int ns,
    f16* __restrict__ G /* [B*R*ns x CINP] */)
{
  int q = blockIdx.x;            // 0 .. B*R-1
  int b = q / RANCH;
  int lane = threadIdx.x;        // 32 threads (one wave)
  __shared__ int sIdx[32];
  __shared__ int sCnt;
  if (lane == 0) sCnt = 0;
  __syncthreads();

  float qx = anchor[(size_t)q * 3 + 0];
  float qy = anchor[(size_t)q * 3 + 1];
  float qz = anchor[(size_t)q * 3 + 2];
  const float* px = xyz + (size_t)b * NPTS * 3;

  for (int base = 0; base < NPTS; base += 32) {
    int cnt = sCnt;
    if (cnt >= ns) break;
    int i = base + lane;
    float dx = px[i * 3 + 0] - qx;
    float dy = px[i * 3 + 1] - qy;
    float dz = px[i * 3 + 2] - qz;
    float d2 = dx * dx + dy * dy + dz * dz;
    bool in = d2 < r2;
    unsigned mask = (unsigned)__ballot(in);
    int pre = __popc(mask & ((1u << lane) - 1u));
    if (in && (cnt + pre) < ns) sIdx[cnt + pre] = i;
    __syncthreads();
    if (lane == 0) {
      int c = sCnt + __popc(mask);
      sCnt = c > ns ? ns : c;
    }
    __syncthreads();
  }

  int cnt = sCnt;
  int v = 0;
  if (lane < ns) v = (cnt == 0) ? 0 : ((lane < cnt) ? sIdx[lane] : sIdx[0]);
  __syncthreads();
  if (lane < ns) sIdx[lane] = v;
  __syncthreads();

  const float* fb = feats + (size_t)b * NPTS * CFEAT;
  for (int s = 0; s < ns; ++s) {
    int pi = sIdx[s];
    f16* row = G + ((size_t)q * ns + s) * CINP;
    const float* pp = px + (size_t)pi * 3;
    const float* ff = fb + (size_t)pi * CFEAT;
    for (int c = lane; c < CINP; c += 32) {
      float val;
      if (c < 3)        val = pp[c] - (c == 0 ? qx : (c == 1 ? qy : qz));
      else if (c < CIN) val = ff[c - 3];
      else              val = 0.f;
      row[c] = (f16)val;
    }
  }
}

// ---------------------------------------------------------------------------
// Ball query + gather for pyramid_level (per-query radius modulated by resid)
// ---------------------------------------------------------------------------
__global__ void k_bq_gather_pyr(
    const float* __restrict__ xyz, const float* __restrict__ feats,
    const float* __restrict__ nxyz, const float* __restrict__ nr,
    const float* __restrict__ resid, int G3, int Mq, float normf,
    f16* __restrict__ G /* [rows*16 x CINP] */,
    f16* __restrict__ GX /* [rows*16 x 32] */,
    unsigned char* __restrict__ emptyB)
{
  int q = blockIdx.x;            // 0 .. B*Mq-1
  int b = q / Mq, m = q - b * Mq;
  int lane = threadIdx.x;
  __shared__ int sIdx[32];
  __shared__ int sCnt;
  if (lane == 0) sCnt = 0;
  __syncthreads();

  float res = resid[(size_t)b * RANCH + (m / G3)];
  float r = nr[q] * (1.f + normf * tanhf(res));
  float r2 = r * r;

  float qx = nxyz[(size_t)q * 3 + 0];
  float qy = nxyz[(size_t)q * 3 + 1];
  float qz = nxyz[(size_t)q * 3 + 2];
  const float* px = xyz + (size_t)b * NPTS * 3;

  for (int base = 0; base < NPTS; base += 32) {
    int cnt = sCnt;
    if (cnt >= NS_PYR) break;
    int i = base + lane;
    float dx = px[i * 3 + 0] - qx;
    float dy = px[i * 3 + 1] - qy;
    float dz = px[i * 3 + 2] - qz;
    float d2 = dx * dx + dy * dy + dz * dz;
    bool in = d2 < r2;
    unsigned mask = (unsigned)__ballot(in);
    int pre = __popc(mask & ((1u << lane) - 1u));
    if (in && (cnt + pre) < NS_PYR) sIdx[cnt + pre] = i;
    __syncthreads();
    if (lane == 0) {
      int c = sCnt + __popc(mask);
      sCnt = c > NS_PYR ? NS_PYR : c;
    }
    __syncthreads();
  }

  int cnt = sCnt;
  if (lane == 0) emptyB[q] = (cnt == 0) ? 1 : 0;
  int v = 0;
  if (lane < NS_PYR) v = (cnt == 0) ? 0 : ((lane < cnt) ? sIdx[lane] : sIdx[0]);
  __syncthreads();
  if (lane < NS_PYR) sIdx[lane] = v;
  __syncthreads();

  const float* fb = feats + (size_t)b * NPTS * CFEAT;
  for (int s = 0; s < NS_PYR; ++s) {
    int pi = sIdx[s];
    f16* row  = G  + ((size_t)q * NS_PYR + s) * CINP;
    f16* rowx = GX + ((size_t)q * NS_PYR + s) * 32;
    const float* pp = px + (size_t)pi * 3;
    const float* ff = fb + (size_t)pi * CFEAT;
    for (int c = lane; c < CINP; c += 32) {
      float val;
      float gx = 0.f;
      if (c < 3) {
        gx = pp[c] - (c == 0 ? qx : (c == 1 ? qy : qz));
        val = gx;
      } else if (c < CIN) val = ff[c - 3];
      else                val = 0.f;
      row[c] = (f16)val;
      if (c < 32) rowx[c] = (f16)(c < 3 ? gx : 0.f);
    }
  }
}

// ---------------------------------------------------------------------------
// Max-pool over ns samples -> pf column block
// ---------------------------------------------------------------------------
__global__ void k_maxpool(const float* __restrict__ H, float* __restrict__ pf,
                          int ns, int colOff)
{
  int idx = blockIdx.x * blockDim.x + threadIdx.x;  // B*R*128
  if (idx >= BATCH * RANCH * DMODEL) return;
  int q = idx >> 7, d = idx & 127;
  float m = -1e30f;
  for (int s = 0; s < ns; ++s) {
    float v = H[((size_t)q * ns + s) * DMODEL + d];
    m = v > m ? v : m;
  }
  pf[(size_t)q * 256 + colOff + d] = m;
}

// ---------------------------------------------------------------------------
// resid = relu(pf @ Wr1^T) @ Wr2^T  (per anchor), 64 threads per row
// ---------------------------------------------------------------------------
__global__ void k_resid(const float* __restrict__ pf,
                        const float* __restrict__ Wr1,
                        const float* __restrict__ Wr2,
                        float* __restrict__ out)
{
  int row = blockIdx.x;          // B*R
  int tid = threadIdx.x;         // 64
  __shared__ float sh[64];
  float a = 0.f;
  const float* pr = pf + (size_t)row * 256;
  const float* wr = Wr1 + (size_t)tid * 256;
  for (int k = 0; k < 256; ++k) a += pr[k] * wr[k];
  a = a > 0.f ? a : 0.f;
  sh[tid] = a * Wr2[tid];
  __syncthreads();
  for (int o = 32; o > 0; o >>= 1) {
    if (tid < o) sh[tid] += sh[tid + o];
    __syncthreads();
  }
  if (tid == 0) out[row] = sh[0];
}

// ---------------------------------------------------------------------------
// Gate tables: gate[g3][d] = sigmoid(pe[g3] . W[d])
// ---------------------------------------------------------------------------
__global__ void k_gate(const float* __restrict__ pe, const float* __restrict__ W,
                       float* __restrict__ gate, int G3)
{
  int idx = blockIdx.x * blockDim.x + threadIdx.x;
  if (idx >= G3 * DMODEL) return;
  int g = idx >> 7, d = idx & 127;
  float a = 0.f;
  for (int k = 0; k < DMODEL; ++k) a += pe[(size_t)g * DMODEL + k] * W[(size_t)d * DMODEL + k];
  gate[idx] = 1.f / (1.f + expf(-a));
}

// ---------------------------------------------------------------------------
// Pyramid attention over ns=16 samples, H=4 heads; output -> Feat f32
// ---------------------------------------------------------------------------
__global__ void k_pyr_attn(
    const f16* __restrict__ Pos, const f16* __restrict__ Key,
    const f16* __restrict__ Val,
    const float* __restrict__ gQK, const float* __restrict__ gK,
    const float* __restrict__ gQ,  const float* __restrict__ gV,
    const float* __restrict__ Wattn, const unsigned char* __restrict__ emptyB,
    float* __restrict__ Feat, int G3)
{
  int q = blockIdx.x;
  int g3 = q % G3;
  int tid = threadIdx.x;         // 128
  __shared__ float sA[NS_PYR * NHEAD];

  const f16* prow = Pos + (size_t)q * NS_PYR * DMODEL;
  const f16* krow = Key + (size_t)q * NS_PYR * DMODEL;
  const f16* vrow = Val + (size_t)q * NS_PYR * DMODEL;
  const float* gqk = gQK + (size_t)g3 * DMODEL;
  const float* gk  = gK  + (size_t)g3 * DMODEL;
  const float* gq  = gQ  + (size_t)g3 * DMODEL;
  const float* gv  = gV  + (size_t)g3 * DMODEL;

  if (tid < NS_PYR * NHEAD) {
    int s = tid >> 2, h = tid & 3;
    float acc = 0.f;
    for (int d = 0; d < DMODEL; ++d) {
      float pv = (float)prow[s * DMODEL + d];
      float kv = (float)krow[s * DMODEL + d];
      acc += (pv * gqk[d] + kv * gk[d]) * Wattn[(size_t)h * DMODEL + d];
    }
    sA[s * NHEAD + h] = acc;
  }
  __syncthreads();
  if (tid < NHEAD) {
    float mx = -1e30f;
    for (int s = 0; s < NS_PYR; ++s) {
      float v = sA[s * NHEAD + tid];
      mx = v > mx ? v : mx;
    }
    float sum = 0.f;
    for (int s = 0; s < NS_PYR; ++s) {
      float e = __expf(sA[s * NHEAD + tid] - mx);
      sA[s * NHEAD + tid] = e;
      sum += e;
    }
    float inv = 1.f / sum;
    for (int s = 0; s < NS_PYR; ++s) sA[s * NHEAD + tid] *= inv;
  }
  __syncthreads();

  int d = tid, h = d >> 5;
  float o = 0.f;
  for (int s = 0; s < NS_PYR; ++s) {
    float vv = (float)vrow[s * DMODEL + d] * gv[d] +
               (float)prow[s * DMODEL + d] * gq[d];
    o += sA[s * NHEAD + h] * vv;
  }
  if (emptyB[q]) o = 0.f;
  Feat[(size_t)q * DMODEL + d] = o;
}

// ---------------------------------------------------------------------------
// Xq = f16(Feat + pe[g3]); X = f16(Feat)
// ---------------------------------------------------------------------------
__global__ void k_make_q(const float* __restrict__ Feat, const float* __restrict__ pe,
                         f16* __restrict__ Xq, f16* __restrict__ X,
                         int G3, int total)
{
  int idx = blockIdx.x * blockDim.x + threadIdx.x;
  if (idx >= total) return;
  int r = idx >> 7, d = idx & 127;
  int g3 = r % G3;
  float x = Feat[idx];
  X[idx]  = (f16)x;
  Xq[idx] = (f16)(x + pe[(size_t)g3 * DMODEL + d]);
}

// ---------------------------------------------------------------------------
// Per-sequence MHA (T <= 64, H=4, HD=32), outputs f16 for the Wo GEMM
// ---------------------------------------------------------------------------
__global__ void k_mha(const f16* __restrict__ Q, const f16* __restrict__ K,
                      const f16* __restrict__ V, f16* __restrict__ O, int T)
{
  int seq = blockIdx.x;
  int tid = threadIdx.x;         // 128
  __shared__ float S[64 * 64];
  const f16* q = Q + (size_t)seq * T * DMODEL;
  const f16* k = K + (size_t)seq * T * DMODEL;
  const f16* v = V + (size_t)seq * T * DMODEL;
  f16* o = O + (size_t)seq * T * DMODEL;
  const float scale = 0.17677669529663687f;   // 1/sqrt(32)

  for (int h = 0; h < NHEAD; ++h) {
    int hb = h * HDIM;
    for (int e = tid; e < T * T; e += 128) {
      int i = e / T, j = e - i * T;
      float a = 0.f;
      for (int d = 0; d < HDIM; ++d)
        a += (float)q[i * DMODEL + hb + d] * (float)k[j * DMODEL + hb + d];
      S[e] = a * scale;
    }
    __syncthreads();
    if (tid < T) {
      float mx = -1e30f;
      for (int j = 0; j < T; ++j) { float x = S[tid * T + j]; mx = x > mx ? x : mx; }
      float sum = 0.f;
      for (int j = 0; j < T; ++j) { float e = __expf(S[tid * T + j] - mx); S[tid * T + j] = e; sum += e; }
      float inv = 1.f / sum;
      for (int j = 0; j < T; ++j) S[tid * T + j] *= inv;
    }
    __syncthreads();
    for (int e = tid; e < T * HDIM; e += 128) {
      int i = e >> 5, d = e & 31;
      float acc = 0.f;
      for (int j = 0; j < T; ++j)
        acc += S[i * T + j] * (float)v[j * DMODEL + hb + d];
      o[i * DMODEL + hb + d] = (f16)acc;
    }
    __syncthreads();
  }
}

// ---------------------------------------------------------------------------
// y = LayerNorm(x + yadd) * gamma + beta;  one block (128 threads) per row
// ---------------------------------------------------------------------------
__global__ void k_res_ln(const float* __restrict__ X, const float* __restrict__ Yadd,
                         const float* __restrict__ gamma, const float* __restrict__ beta,
                         float* __restrict__ Out, f16* __restrict__ Outh)
{
  int row = blockIdx.x;
  int tid = threadIdx.x;         // 128
  __shared__ float red[128];
  size_t idx = (size_t)row * DMODEL + tid;
  float v = X[idx] + Yadd[idx];
  red[tid] = v;
  __syncthreads();
  for (int o = 64; o > 0; o >>= 1) { if (tid < o) red[tid] += red[tid + o]; __syncthreads(); }
  float mean = red[0] * (1.f / DMODEL);
  __syncthreads();
  float dv = v - mean;
  red[tid] = dv * dv;
  __syncthreads();
  for (int o = 64; o > 0; o >>= 1) { if (tid < o) red[tid] += red[tid + o]; __syncthreads(); }
  float var = red[0] * (1.f / DMODEL);
  float y = dv * rsqrtf(var + LN_EPS) * gamma[tid] + beta[tid];
  Out[idx] = y;
  if (Outh) Outh[idx] = (f16)y;
}

// ---------------------------------------------------------------------------
// weight_pool: w = sigmoid(relu(X@W1^T + b1)@W2 + b2); out = sum_t X*w
// One block (128 threads) per sequence.
// ---------------------------------------------------------------------------
__global__ void k_pool(const float* __restrict__ X,
                       const float* __restrict__ W1, const float* __restrict__ b1,
                       const float* __restrict__ W2, const float* __restrict__ b2,
                       float* __restrict__ out, int T, int ldout, int colOff)
{
  int seq = blockIdx.x;
  int tid = threadIdx.x;         // 128
  __shared__ float sh[64 * 64];
  __shared__ float sw[64];
  const float* x = X + (size_t)seq * T * DMODEL;

  for (int e = tid; e < T * 64; e += 128) {
    int t = e >> 6, j = e & 63;
    float a = b1[j];
    const float* w = W1 + (size_t)j * DMODEL;
    const float* xr = x + (size_t)t * DMODEL;
    for (int k = 0; k < DMODEL; ++k) a += xr[k] * w[k];
    sh[e] = a > 0.f ? a : 0.f;
  }
  __syncthreads();
  if (tid < T) {
    float a = b2[0];
    for (int j = 0; j < 64; ++j) a += sh[tid * 64 + j] * W2[j];
    sw[tid] = 1.f / (1.f + expf(-a));
  }
  __syncthreads();
  float acc = 0.f;
  for (int t = 0; t < T; ++t) acc += x[(size_t)t * DMODEL + tid] * sw[t];
  out[(size_t)seq * ldout + colOff + tid] = acc;
}

// ---------------------------------------------------------------------------
// Host orchestration
// ---------------------------------------------------------------------------
struct Bump {
  char* base; size_t off;
  void* take(size_t n) {
    size_t a = (off + 255) & ~(size_t)255;
    off = a + n;
    return base + a;
  }
};

static void gemm(hipStream_t s, const f16* A, int lda, const f16* Wt, int ldw,
                 const float* bias, f16* Ch, float* Cf, int ldc,
                 int M, int K, int act)
{
  int strips = M / 16;
  int blocks = (strips + 7) / 8;                     // 8 waves per block
  size_t smem = (size_t)DMODEL * (K + LDS_PAD) * sizeof(f16);
  k_wmma_gemm128<<<blocks, 256, smem, s>>>(A, lda, Wt, ldw, bias, Ch, Cf, ldc, M, K, act);
}

static void cvt(hipStream_t s, const float* src, f16* dst, int Nrows, int K, int Kp)
{
  int total = Nrows * Kp;
  k_cvt<<<(total + 255) / 256, 256, 0, s>>>(src, dst, Nrows, K, Kp);
}

extern "C" void kernel_launch(void* const* d_in, const int* in_sizes, int n_in,
                              void* d_out, int out_size, void* d_ws, size_t ws_size,
                              hipStream_t stream)
{
  (void)in_sizes; (void)n_in; (void)out_size; (void)ws_size;
  const float* xyz    = (const float*)d_in[I_XYZ];
  const float* feats  = (const float*)d_in[I_FEAT];
  const float* anchor = (const float*)d_in[I_ANCHOR];
  const float* nxyz[2] = { (const float*)d_in[I_NX0], (const float*)d_in[I_NX1] };
  const float* nrad[2] = { (const float*)d_in[I_NR0], (const float*)d_in[I_NR1] };
  float* out_cls = (float*)d_out;                 // (256, 256)
  float* out_reg = (float*)d_out + 256 * 256;     // (256, 256)

  const int   G3v[2]    = { 27, 64 };
  const float predR2[2] = { 0.8f * 0.8f, 1.6f * 1.6f };
  const int   predNS[2] = { 16, 32 };

  Bump ws { (char*)d_ws, 0 };
  float* pf = (float*)ws.take((size_t)BATCH * RANCH * 256 * sizeof(float));
  size_t mark = ws.off;

  // ---------------- predict levels -> pf (B*R x 256) ----------------
  for (int j = 0; j < 2; ++j) {
    ws.off = mark;
    int ns = predNS[j];
    int rows = BATCH * RANCH * ns;
    const float* W1 = (const float*)d_in[j == 0 ? I_P0W1 : I_P1W1];
    const float* W2 = (const float*)d_in[j == 0 ? I_P0W2 : I_P1W2];
    f16* W1h = (f16*)ws.take((size_t)DMODEL * CINP * sizeof(f16));
    f16* W2h = (f16*)ws.take((size_t)DMODEL * DMODEL * sizeof(f16));
    cvt(stream, W1, W1h, DMODEL, CIN, CINP);
    cvt(stream, W2, W2h, DMODEL, DMODEL, DMODEL);

    f16*   Gp = (f16*)ws.take((size_t)rows * CINP * sizeof(f16));
    f16*   H1 = (f16*)ws.take((size_t)rows * DMODEL * sizeof(f16));
    float* H2 = (float*)ws.take((size_t)rows * DMODEL * sizeof(float));

    k_bq_gather_pred<<<BATCH * RANCH, 32, 0, stream>>>(xyz, feats, anchor,
                                                       predR2[j], ns, Gp);
    gemm(stream, Gp, CINP, W1h, CINP, nullptr, H1, nullptr, DMODEL, rows, CINP, 1);
    gemm(stream, H1, DMODEL, W2h, DMODEL, nullptr, nullptr, H2, DMODEL, rows, DMODEL, 1);
    int tot = BATCH * RANCH * DMODEL;
    k_maxpool<<<(tot + 255) / 256, 256, 0, stream>>>(H2, pf, ns, j * DMODEL);
  }

  // ---------------- pyramid levels ----------------
  for (int i = 0; i < 2; ++i) {
    ws.off = mark;
    int G3 = G3v[i];
    int Mq = RANCH * G3;               // queries per batch
    int rows = BATCH * Mq;             // total tokens
    int srows = rows * NS_PYR;         // total gathered samples
    const int base = I_PYR0 + 32 * i;
#define PP(k) ((const float*)d_in[base + (k)])

    // f16 weights
    f16* Wposh = (f16*)ws.take((size_t)DMODEL * 32 * sizeof(f16));
    f16* Wkeyh = (f16*)ws.take((size_t)DMODEL * CINP * sizeof(f16));
    f16* Wv1h  = (f16*)ws.take((size_t)DMODEL * CINP * sizeof(f16));
    f16* Wv2h  = (f16*)ws.take((size_t)DMODEL * DMODEL * sizeof(f16));
    f16* Wqkvh = (f16*)ws.take((size_t)3 * DMODEL * DMODEL * sizeof(f16));
    f16* Woh   = (f16*)ws.take((size_t)DMODEL * DMODEL * sizeof(f16));
    f16* W1h   = (f16*)ws.take((size_t)DMODEL * DMODEL * sizeof(f16));
    f16* W2h   = (f16*)ws.take((size_t)DMODEL * DMODEL * sizeof(f16));
    cvt(stream, PP(P_WPOS), Wposh, DMODEL, 3, 32);
    cvt(stream, PP(P_WKEY), Wkeyh, DMODEL, CIN, CINP);
    cvt(stream, PP(P_WV1),  Wv1h,  DMODEL, CIN, CINP);
    cvt(stream, PP(P_WV2),  Wv2h,  DMODEL, DMODEL, DMODEL);
    cvt(stream, PP(P_WQKV), Wqkvh, 3 * DMODEL, DMODEL, DMODEL);
    cvt(stream, PP(P_WO),   Woh,   DMODEL, DMODEL, DMODEL);
    cvt(stream, PP(P_W1),   W1h,   DMODEL, DMODEL, DMODEL);
    cvt(stream, PP(P_W2),   W2h,   DMODEL, DMODEL, DMODEL);

    // gate tables
    float* gQK = (float*)ws.take((size_t)G3 * DMODEL * sizeof(float));
    float* gK  = (float*)ws.take((size_t)G3 * DMODEL * sizeof(float));
    float* gQ  = (float*)ws.take((size_t)G3 * DMODEL * sizeof(float));
    float* gV  = (float*)ws.take((size_t)G3 * DMODEL * sizeof(float));
    int gt = G3 * DMODEL;
    k_gate<<<(gt + 255) / 256, 256, 0, stream>>>(PP(P_PE), PP(P_WQKC), gQK, G3);
    k_gate<<<(gt + 255) / 256, 256, 0, stream>>>(PP(P_PE), PP(P_WKC),  gK,  G3);
    k_gate<<<(gt + 255) / 256, 256, 0, stream>>>(PP(P_PE), PP(P_WQC),  gQ,  G3);
    k_gate<<<(gt + 255) / 256, 256, 0, stream>>>(PP(P_PE), PP(P_WVC),  gV,  G3);

    // residual radius modulation
    float* residB = (float*)ws.take((size_t)BATCH * RANCH * sizeof(float));
    k_resid<<<BATCH * RANCH, 64, 0, stream>>>(pf, PP(P_WR1), PP(P_WR2), residB);

    // gather + point MLPs (the FLOP-dominant WMMA GEMMs)
    f16* Gb = (f16*)ws.take((size_t)srows * CINP * sizeof(f16));
    f16* GX = (f16*)ws.take((size_t)srows * 32 * sizeof(f16));
    unsigned char* emptyB = (unsigned char*)ws.take(rows);
    k_bq_gather_pyr<<<rows, 32, 0, stream>>>(xyz, feats, nxyz[i], nrad[i],
                                             residB, G3, Mq, 0.5f, Gb, GX, emptyB);

    f16* Pos = (f16*)ws.take((size_t)srows * DMODEL * sizeof(f16));
    f16* Key = (f16*)ws.take((size_t)srows * DMODEL * sizeof(f16));
    f16* Tmp = (f16*)ws.take((size_t)srows * DMODEL * sizeof(f16));
    f16* Val = (f16*)ws.take((size_t)srows * DMODEL * sizeof(f16));
    gemm(stream, GX, 32,   Wposh, 32,   nullptr, Pos, nullptr, DMODEL, srows, 32,   1);
    gemm(stream, Gb, CINP, Wkeyh, CINP, nullptr, Key, nullptr, DMODEL, srows, CINP, 1);
    gemm(stream, Gb, CINP, Wv1h,  CINP, nullptr, Tmp, nullptr, DMODEL, srows, CINP, 1);
    gemm(stream, Tmp, DMODEL, Wv2h, DMODEL, nullptr, Val, nullptr, DMODEL, srows, DMODEL, 1);

    float* Feat = (float*)ws.take((size_t)rows * DMODEL * sizeof(float));
    k_pyr_attn<<<rows, 128, 0, stream>>>(Pos, Key, Val, gQK, gK, gQ, gV,
                                         PP(P_WATTN), emptyB, Feat, G3);

    // -------- encoder layer --------
    f16* Xq16 = (f16*)ws.take((size_t)rows * DMODEL * sizeof(f16));
    f16* X16  = (f16*)ws.take((size_t)rows * DMODEL * sizeof(f16));
    int tot = rows * DMODEL;
    k_make_q<<<(tot + 255) / 256, 256, 0, stream>>>(Feat, PP(P_PE), Xq16, X16, G3, tot);

    f16* Qb = (f16*)ws.take((size_t)rows * DMODEL * sizeof(f16));
    f16* Kb = (f16*)ws.take((size_t)rows * DMODEL * sizeof(f16));
    f16* Vb = (f16*)ws.take((size_t)rows * DMODEL * sizeof(f16));
    const float* bqkv = PP(P_BQKV);
    gemm(stream, Xq16, DMODEL, Wqkvh,                       DMODEL, bqkv,       Qb, nullptr, DMODEL, rows, DMODEL, 0);
    gemm(stream, Xq16, DMODEL, Wqkvh + 1 * DMODEL * DMODEL, DMODEL, bqkv + 128, Kb, nullptr, DMODEL, rows, DMODEL, 0);
    gemm(stream, X16,  DMODEL, Wqkvh + 2 * DMODEL * DMODEL, DMODEL, bqkv + 256, Vb, nullptr, DMODEL, rows, DMODEL, 0);

    int nseq = rows / G3;   // = 256
    f16* O16 = (f16*)ws.take((size_t)rows * DMODEL * sizeof(f16));
    k_mha<<<nseq, 128, 0, stream>>>(Qb, Kb, Vb, O16, G3);

    float* Obuf = (float*)ws.take((size_t)rows * DMODEL * sizeof(float));
    gemm(stream, O16, DMODEL, Woh, DMODEL, PP(P_BO), nullptr, Obuf, DMODEL, rows, DMODEL, 0);

    float* X2  = (float*)ws.take((size_t)rows * DMODEL * sizeof(float));
    f16*   X2h = (f16*)ws.take((size_t)rows * DMODEL * sizeof(f16));
    k_res_ln<<<rows, 128, 0, stream>>>(Feat, Obuf, PP(P_LN1G), PP(P_LN1B), X2, X2h);

    f16*   F1 = (f16*)ws.take((size_t)rows * DMODEL * sizeof(f16));
    float* F2 = (float*)ws.take((size_t)rows * DMODEL * sizeof(float));
    gemm(stream, X2h, DMODEL, W1h, DMODEL, PP(P_B1), F1, nullptr, DMODEL, rows, DMODEL, 1);
    gemm(stream, F1,  DMODEL, W2h, DMODEL, PP(P_B2), nullptr, F2, DMODEL, rows, DMODEL, 0);

    float* X3 = (float*)ws.take((size_t)rows * DMODEL * sizeof(float));
    k_res_ln<<<rows, 128, 0, stream>>>(X2, F2, PP(P_LN2G), PP(P_LN2B), X3, nullptr);

    // -------- pooled heads into d_out --------
    k_pool<<<nseq, 128, 0, stream>>>(X3, PP(P_CW1), PP(P_CB1), PP(P_CW2), PP(P_CB2),
                                     out_cls, G3, 256, i * DMODEL);
    k_pool<<<nseq, 128, 0, stream>>>(X3, PP(P_RW1), PP(P_RB1), PP(P_RW2), PP(P_RB2),
                                     out_reg, G3, 256, i * DMODEL);
#undef PP
  }
}
